// SAttention_81192061764136
// MI455X (gfx1250) — matrix operand order repaired and checked by reading, not compiled, not run on hardware
//
#include <hip/hip_runtime.h>
#include <hip/hip_fp16.h>
#include <math.h>

// ---------------------------------------------------------------------------
// Transformer encoder layer forward for gfx1250 (MI455X, wave32, WMMA).
// GEMMs: v_wmma_f32_16x16x32_f16 with double-buffered LDS tiles filled by
// global_load_async_to_lds_b128 (ASYNCcnt path), fragments read via ds_load.
// ---------------------------------------------------------------------------

typedef __attribute__((ext_vector_type(16))) _Float16 v16h;
typedef __attribute__((ext_vector_type(8)))  _Float16 v8h;
typedef __attribute__((ext_vector_type(8)))  float    v8f;

#define WMMA_F16(a, b, c) \
    __builtin_amdgcn_wmma_f32_16x16x32_f16(false, (a), false, (b), (short)0, (c), false, false)

__device__ __forceinline__ int lane_id() { return threadIdx.x & 31; }

// LDS offset (wave-relative byte address) of a __shared__ object: low 32 bits
// of the generic address (aperture check discards the upper bits, ISA 10.2).
__device__ __forceinline__ unsigned lds_off(const void* p) {
    return (unsigned)(unsigned long long)(uintptr_t)p;
}

// ---------------------------------------------------------------------------
// Fragment loads (global)
// ---------------------------------------------------------------------------
// A-fragment: 16x32 f16, rows m0..m0+15, K = kb..kb+31, row-major src.
// lanes 0-15 hold K 0..15, lanes 16-31 hold K 16..31.
__device__ __forceinline__ v16h load_a_frag(const _Float16* __restrict__ base, int lda,
                                            int m0, int kb, int lane) {
    int row  = m0 + (lane & 15);
    int koff = (lane < 16) ? 0 : 16;
    const _Float16* p = base + (long)row * lda + kb + koff;
    v8h lo = *(const v8h*)(p);
    v8h hi = *(const v8h*)(p + 8);
    v16h r;
#pragma unroll
    for (int i = 0; i < 8; ++i) { r[i] = lo[i]; r[i + 8] = hi[i]; }
    return r;
}

// B-fragment: 32x16 f16 (KxN) from pre-transposed Bt[N,K] row-major.
// lane holds column n=lane&15; lanes 0-15 K 0..15, lanes 16-31 K 16..31.
__device__ __forceinline__ v16h load_b_frag(const _Float16* __restrict__ base, int ldb,
                                            int n0, int kb, int lane) {
    int row  = n0 + (lane & 15);
    int koff = (lane < 16) ? 0 : 16;
    return *(const v16h*)(base + (long)row * ldb + kb + koff);
}

// ---------------------------------------------------------------------------
// Async copy of a 128-row x 32-half tile into LDS (padded stride 40 halves).
// 512 x 16B chunks, 256 threads -> 2 async b128 copies per thread.
// ---------------------------------------------------------------------------
#define LDSK   40            // halves per LDS tile row (32 data + 8 pad)
#define TILEH  (128 * LDSK)  // halves per tile buffer

__device__ __forceinline__ void issue_tile_copy(const _Float16* __restrict__ g, int ldk,
                                                int row0, int kb, unsigned lds_base, int tid) {
#pragma unroll
    for (int t = 0; t < 2; ++t) {
        int chunk = tid + t * 256;
        int r = chunk >> 2, c = chunk & 3;
        unsigned lds = lds_base + (unsigned)(r * LDSK + c * 8) * 2u;
        unsigned long long gp =
            (unsigned long long)(uintptr_t)(g + (long)(row0 + r) * ldk + kb + c * 8);
        asm volatile("global_load_async_to_lds_b128 %0, %1, off"
                     :: "v"(lds), "v"(gp) : "memory");
    }
}

__device__ __forceinline__ void wait_async_all() {
    asm volatile("s_wait_asynccnt 0" ::: "memory");
}

// ---------------------------------------------------------------------------
// One k-step: 2 A-frags + 4 B-frags from LDS (ds_load), 8 WMMAs.
// pa/pb resolve to direct __shared__ bases after inlining -> ds_load_b128.
// ---------------------------------------------------------------------------
__device__ __forceinline__ void compute_step(const _Float16* pa, const _Float16* pb,
                                             int lane, int wm, int wn, v8f acc[2][4]) {
    v16h af[2];
#pragma unroll
    for (int rf = 0; rf < 2; ++rf) {
        int row  = wm + rf * 16 + (lane & 15);
        int koff = (lane < 16) ? 0 : 16;
        const _Float16* p = pa + row * LDSK + koff;
        v8h lo = *(const v8h*)(p);
        v8h hi = *(const v8h*)(p + 8);
#pragma unroll
        for (int e = 0; e < 8; ++e) { af[rf][e] = lo[e]; af[rf][e + 8] = hi[e]; }
    }
    v16h bf[4];
#pragma unroll
    for (int cf = 0; cf < 4; ++cf) {
        int col  = wn + cf * 16 + (lane & 15);
        int koff = (lane < 16) ? 0 : 16;
        bf[cf] = *(const v16h*)(pb + col * LDSK + koff);
    }
#pragma unroll
    for (int rf = 0; rf < 2; ++rf)
#pragma unroll
        for (int cf = 0; cf < 4; ++cf)
            acc[rf][cf] = WMMA_F16(af[rf], bf[cf], acc[rf][cf]);
}

// ---------------------------------------------------------------------------
// WMMA GEMM core, double-buffered LDS, manual unroll-2 (constant buffer ids).
// Block = 256 threads = 8 waves, 4(M) x 2(N); wave tile 32x64 -> block 128x128.
// ---------------------------------------------------------------------------
template <int KK>
__device__ __forceinline__ void gemm_core_lds(const _Float16* __restrict__ A,
                                              const _Float16* __restrict__ Bt,
                                              int m0blk, int n0blk,
                                              _Float16 (&sA)[2][TILEH],
                                              _Float16 (&sB)[2][TILEH],
                                              int tid, int lane, int wm, int wn,
                                              v8f acc[2][4]) {
    unsigned oA0 = lds_off(&sA[0][0]), oA1 = lds_off(&sA[1][0]);
    unsigned oB0 = lds_off(&sB[0][0]), oB1 = lds_off(&sB[1][0]);

    issue_tile_copy(A,  KK, m0blk, 0, oA0, tid);
    issue_tile_copy(Bt, KK, n0blk, 0, oB0, tid);

    const int nk = KK / 32;  // even
#pragma unroll 1
    for (int i = 0; i < nk; i += 2) {
        wait_async_all();
        __syncthreads();
        issue_tile_copy(A,  KK, m0blk, (i + 1) * 32, oA1, tid);
        issue_tile_copy(Bt, KK, n0blk, (i + 1) * 32, oB1, tid);
        compute_step(&sA[0][0], &sB[0][0], lane, wm, wn, acc);

        wait_async_all();
        __syncthreads();
        if (i + 2 < nk) {
            issue_tile_copy(A,  KK, m0blk, (i + 2) * 32, oA0, tid);
            issue_tile_copy(Bt, KK, n0blk, (i + 2) * 32, oB0, tid);
        }
        compute_step(&sA[1][0], &sB[1][0], lane, wm, wn, acc);
    }
}

#define GEMM_PREAMBLE(KVAL)                                            \
    __shared__ __align__(32) _Float16 sA[2][TILEH];                    \
    __shared__ __align__(32) _Float16 sB[2][TILEH];                    \
    int tid = threadIdx.x, lane = tid & 31, w = tid >> 5;              \
    int wm = (w >> 1) * 32, wn = (w & 1) * 64;                         \
    int m0blk = blockIdx.x * 128, n0blk = blockIdx.y * 128;            \
    v8f acc[2][4] = {};                                                \
    gemm_core_lds<KVAL>(A16, Wt, m0blk, n0blk, sA, sB,                 \
                        tid, lane, wm, wn, acc);                       \
    int colw = lane & 15, rbase = (lane < 16) ? 0 : 8;

// QKV projection: rows r = a*16 + b of xln16, cols c = h*64 + dh.
// vmode==0 -> out[((b*8+h)*1024 + a)*64 + dh]   (Q or K, [B,H,A,DH])
// vmode==1 -> out[((b*8+h)*64 + dh)*1024 + a]   (V transposed, [B,H,DH,A])
__global__ void __launch_bounds__(256)
gemm_qkv_kernel(const _Float16* __restrict__ A16, const _Float16* __restrict__ Wt,
                const float* __restrict__ bias, _Float16* __restrict__ out, int vmode) {
    GEMM_PREAMBLE(512)
#pragma unroll
    for (int rf = 0; rf < 2; ++rf)
#pragma unroll
        for (int cf = 0; cf < 4; ++cf)
#pragma unroll
            for (int r = 0; r < 8; ++r) {
                int row = m0blk + wm + rf * 16 + rbase + r;
                int a = row >> 4, b = row & 15;
                int c = n0blk + wn + cf * 16 + colw;
                float val = acc[rf][cf][r] + bias[c];
                int h = c >> 6, dh = c & 63;
                long idx = vmode ? (((long)(b * 8 + h) * 64 + dh) * 1024 + a)
                                 : (((long)(b * 8 + h) * 1024 + a) * 64 + dh);
                out[idx] = (_Float16)val;
            }
}

// FFN1: h = gelu_exact(x @ W1 + b1), output f16 [M,1024]
__global__ void __launch_bounds__(256)
gemm_ffn1_kernel(const _Float16* __restrict__ A16, const _Float16* __restrict__ Wt,
                 const float* __restrict__ bias, _Float16* __restrict__ h16) {
    GEMM_PREAMBLE(512)
    const int N = 1024;
#pragma unroll
    for (int rf = 0; rf < 2; ++rf)
#pragma unroll
        for (int cf = 0; cf < 4; ++cf)
#pragma unroll
            for (int r = 0; r < 8; ++r) {
                int row = m0blk + wm + rf * 16 + rbase + r;
                int c   = n0blk + wn + cf * 16 + colw;
                float x = acc[rf][cf][r] + bias[c];
                float g = 0.5f * x * (1.0f + erff(x * 0.70710678118654752f));
                h16[(long)row * N + c] = (_Float16)g;
            }
}

// FFN2: out = xt + h @ W2 + b2, output f32 [M,512]
__global__ void __launch_bounds__(256)
gemm_ffn2_kernel(const _Float16* __restrict__ A16, const _Float16* __restrict__ Wt,
                 const float* __restrict__ bias, const float* __restrict__ xt32,
                 float* __restrict__ out) {
    GEMM_PREAMBLE(1024)
    const int N = 512;
#pragma unroll
    for (int rf = 0; rf < 2; ++rf)
#pragma unroll
        for (int cf = 0; cf < 4; ++cf)
#pragma unroll
            for (int r = 0; r < 8; ++r) {
                int  row = m0blk + wm + rf * 16 + rbase + r;
                int  c   = n0blk + wn + cf * 16 + colw;
                long idx = (long)row * N + c;
                out[idx] = xt32[idx] + acc[rf][cf][r] + bias[c];
            }
}

// ---------------------------------------------------------------------------
// Weight convert + transpose: Wt[n*K + k] = (f16) W[k*N + n]
// ---------------------------------------------------------------------------
__global__ void w_transpose_f16(const float* __restrict__ W, _Float16* __restrict__ Wt,
                                int K, int N) {
    long i = (long)blockIdx.x * 256 + threadIdx.x;
    if (i < (long)K * N) {
        int n = (int)(i / K);
        int k = (int)(i % K);
        Wt[i] = (_Float16)W[(long)k * N + n];
    }
}

// ---------------------------------------------------------------------------
// LayerNorm over d=512, optional residual add. Writes f32 and f16 copies.
// ---------------------------------------------------------------------------
__global__ void ln_kernel(const float* __restrict__ xin, const float* __restrict__ addin,
                          const float* __restrict__ g, const float* __restrict__ be,
                          float* __restrict__ o32, _Float16* __restrict__ o16) {
    int  t    = threadIdx.x;
    long base = (long)blockIdx.x * 512;
    float v0 = xin[base + t], v1 = xin[base + t + 256];
    if (addin) { v0 += addin[base + t]; v1 += addin[base + t + 256]; }
    float s = v0 + v1, q = v0 * v0 + v1 * v1;
#pragma unroll
    for (int off = 16; off >= 1; off >>= 1) {
        s += __shfl_xor(s, off, 32);
        q += __shfl_xor(q, off, 32);
    }
    __shared__ float sh[2][8];
    int lane = t & 31, w = t >> 5;
    if (lane == 0) { sh[0][w] = s; sh[1][w] = q; }
    __syncthreads();
    if (t == 0) {
        float ts = 0.f, tq = 0.f;
#pragma unroll
        for (int i = 0; i < 8; ++i) { ts += sh[0][i]; tq += sh[1][i]; }
        float mean = ts * (1.f / 512.f);
        sh[0][0] = mean;
        sh[1][0] = tq * (1.f / 512.f) - mean * mean;  // jnp.var, ddof=0
    }
    __syncthreads();
    float mean = sh[0][0];
    float rstd = rsqrtf(sh[1][0] + 1e-5f);
    float y0 = (v0 - mean) * rstd * g[t] + be[t];
    float y1 = (v1 - mean) * rstd * g[t + 256] + be[t + 256];
    o32[base + t]       = y0;
    o32[base + t + 256] = y1;
    o16[base + t]       = (_Float16)y0;
    o16[base + t + 256] = (_Float16)y1;
}

// ---------------------------------------------------------------------------
// Attention pass 1: per-head sum / sumsq of raw scores (for the global std).
// ---------------------------------------------------------------------------
__global__ void attn_stats_kernel(const _Float16* __restrict__ q16,
                                  const _Float16* __restrict__ k16,
                                  float* __restrict__ stats) {
    const int A = 1024, DH = 64;
    int lane = lane_id(), wid = threadIdx.x >> 5;
    int bh = blockIdx.y;
    int h  = bh & 7;
    int m0 = (blockIdx.x * 4 + wid) * 16;
    const _Float16* q = q16 + (long)bh * A * DH;
    const _Float16* k = k16 + (long)bh * A * DH;
    v16h aq0 = load_a_frag(q, DH, m0, 0, lane);
    v16h aq1 = load_a_frag(q, DH, m0, 32, lane);
    float s1 = 0.f, s2 = 0.f;
    for (int n0 = 0; n0 < A; n0 += 16) {
        v16h b0 = load_b_frag(k, DH, n0, 0, lane);
        v16h b1 = load_b_frag(k, DH, n0, 32, lane);
        v8f s = {};
        s = WMMA_F16(aq0, b0, s);
        s = WMMA_F16(aq1, b1, s);
#pragma unroll
        for (int r = 0; r < 8; ++r) { s1 += s[r]; s2 += s[r] * s[r]; }
    }
#pragma unroll
    for (int off = 16; off >= 1; off >>= 1) {
        s1 += __shfl_xor(s1, off, 32);
        s2 += __shfl_xor(s2, off, 32);
    }
    if (lane == 0) {
        atomicAdd(&stats[h * 2 + 0], s1);
        atomicAdd(&stats[h * 2 + 1], s2);
    }
}

// scaling[h] = TEMP * sqrt(1 + std); store reciprocal. ddof=1 over N=B*A*A.
__global__ void attn_scale_kernel(const float* __restrict__ stats,
                                  float* __restrict__ invscale) {
    int h = threadIdx.x;
    if (h < 8) {
        const float N  = 16.f * 1024.f * 1024.f;
        float s1 = stats[h * 2], s2 = stats[h * 2 + 1];
        float var = (s2 - s1 * s1 / N) / (N - 1.f);
        float sd  = sqrtf(fmaxf(var, 0.f));
        invscale[h] = 1.f / (8.f * sqrtf(1.f + sd));  // TEMP = sqrt(512/8) = 8
    }
}

// ---------------------------------------------------------------------------
// Attention pass 2: softmax(S/scale) @ V, one wave per (b,h,query-tile).
// ---------------------------------------------------------------------------
__global__ void attn_softmax_pv_kernel(const _Float16* __restrict__ q16,
                                       const _Float16* __restrict__ k16,
                                       const _Float16* __restrict__ vT16,
                                       const float* __restrict__ invscale,
                                       float* __restrict__ attn_out) {
    const int A = 1024, DH = 64;
    __shared__ __align__(32) _Float16 pbuf[4][16 * 32];
    int lane = lane_id(), wid = threadIdx.x >> 5;
    int bh = blockIdx.y;
    int b = bh >> 3, h = bh & 7;
    int m0 = (blockIdx.x * 4 + wid) * 16;
    const _Float16* q  = q16 + (long)bh * A * DH;
    const _Float16* k  = k16 + (long)bh * A * DH;
    const _Float16* vT = vT16 + (long)bh * DH * A;  // [DH, A]
    float inv = invscale[h];

    v16h aq0 = load_a_frag(q, DH, m0, 0, lane);
    v16h aq1 = load_a_frag(q, DH, m0, 32, lane);

    int colw = lane & 15, rbase = (lane < 16) ? 0 : 8;

    // ---- stream 1: row maxima ----
    float mx[8];
#pragma unroll
    for (int r = 0; r < 8; ++r) mx[r] = -3.0e38f;
    for (int n0 = 0; n0 < A; n0 += 16) {
        v16h b0 = load_b_frag(k, DH, n0, 0, lane);
        v16h b1 = load_b_frag(k, DH, n0, 32, lane);
        v8f s = {};
        s = WMMA_F16(aq0, b0, s);
        s = WMMA_F16(aq1, b1, s);
#pragma unroll
        for (int r = 0; r < 8; ++r) mx[r] = fmaxf(mx[r], s[r]);
    }
#pragma unroll
    for (int r = 0; r < 8; ++r) {
#pragma unroll
        for (int off = 8; off >= 1; off >>= 1)
            mx[r] = fmaxf(mx[r], __shfl_xor(mx[r], off, 32));
        mx[r] *= inv;
    }

    // ---- stream 2: P = exp(S*inv - mx), O += P @ V ----
    float rs[8];
#pragma unroll
    for (int r = 0; r < 8; ++r) rs[r] = 0.f;
    v8f o0 = {}, o1 = {}, o2 = {}, o3 = {};
    _Float16* pb = pbuf[wid];

    for (int kb2 = 0; kb2 < A; kb2 += 32) {
        v8f slo = {}, shi = {};
        {
            v16h b0 = load_b_frag(k, DH, kb2, 0, lane);
            v16h b1 = load_b_frag(k, DH, kb2, 32, lane);
            slo = WMMA_F16(aq0, b0, slo);
            slo = WMMA_F16(aq1, b1, slo);
        }
        {
            v16h b0 = load_b_frag(k, DH, kb2 + 16, 0, lane);
            v16h b1 = load_b_frag(k, DH, kb2 + 16, 32, lane);
            shi = WMMA_F16(aq0, b0, shi);
            shi = WMMA_F16(aq1, b1, shi);
        }
#pragma unroll
        for (int r = 0; r < 8; ++r) {
            float plo = __expf(slo[r] * inv - mx[r]);
            float phi = __expf(shi[r] * inv - mx[r]);
            rs[r] += plo + phi;
            int row = rbase + r;
            pb[row * 32 + colw]      = (_Float16)plo;
            pb[row * 32 + 16 + colw] = (_Float16)phi;
        }
        // re-read P as an A-fragment (16 rows x 32 keys)
        int prow = lane & 15, pk = (lane < 16) ? 0 : 16;
        v8h lo = *(const v8h*)&pb[prow * 32 + pk];
        v8h hi = *(const v8h*)&pb[prow * 32 + 8 + pk];
        v16h pa;
#pragma unroll
        for (int i = 0; i < 8; ++i) { pa[i] = lo[i]; pa[i + 8] = hi[i]; }

        o0 = WMMA_F16(pa, load_b_frag(vT, A, 0,  kb2, lane), o0);
        o1 = WMMA_F16(pa, load_b_frag(vT, A, 16, kb2, lane), o1);
        o2 = WMMA_F16(pa, load_b_frag(vT, A, 32, kb2, lane), o2);
        o3 = WMMA_F16(pa, load_b_frag(vT, A, 48, kb2, lane), o3);
    }

#pragma unroll
    for (int r = 0; r < 8; ++r) {
#pragma unroll
        for (int off = 8; off >= 1; off >>= 1) rs[r] += __shfl_xor(rs[r], off, 32);
    }
#pragma unroll
    for (int r = 0; r < 8; ++r) {
        float invsum = 1.f / rs[r];
        int   a      = m0 + rbase + r;
        float* orow  = attn_out + ((long)a * 16 + b) * 512 + h * 64;
        orow[colw]      = o0[r] * invsum;
        orow[16 + colw] = o1[r] * invsum;
        orow[32 + colw] = o2[r] * invsum;
        orow[48 + colw] = o3[r] * invsum;
    }
}

// ---------------------------------------------------------------------------
// Host launch
// ---------------------------------------------------------------------------
extern "C" void kernel_launch(void* const* d_in, const int* in_sizes, int n_in,
                              void* d_out, int out_size, void* d_ws, size_t ws_size,
                              hipStream_t stream) {
    const int A = 1024, B = 16, D = 512, F = 1024, H = 8, DH = 64;
    const int M = A * B;  // 16384 rows

    const float* x   = (const float*)d_in[0];
    const float* Wq  = (const float*)d_in[1];
    const float* bq  = (const float*)d_in[2];
    const float* Wk  = (const float*)d_in[3];
    const float* bk  = (const float*)d_in[4];
    const float* Wv  = (const float*)d_in[5];
    const float* bv  = (const float*)d_in[6];
    const float* g1  = (const float*)d_in[7];
    const float* be1 = (const float*)d_in[8];
    const float* g2  = (const float*)d_in[9];
    const float* be2 = (const float*)d_in[10];
    const float* W1  = (const float*)d_in[11];
    const float* bf1 = (const float*)d_in[12];
    const float* W2  = (const float*)d_in[13];
    const float* bf2 = (const float*)d_in[14];
    float* out = (float*)d_out;

    char* ws = (char*)d_ws;
    size_t off = 0;
    auto carve = [&](size_t bytes) { void* p = ws + off; off += (bytes + 255) & ~(size_t)255; return p; };
    float*     xln32 = (float*)    carve((size_t)M * D * 4);
    _Float16*  xln16 = (_Float16*) carve((size_t)M * D * 2);
    float*     xt32  = (float*)    carve((size_t)M * D * 4);
    _Float16*  xt16  = (_Float16*) carve((size_t)M * D * 2);
    float*     attn  = (float*)    carve((size_t)M * D * 4);
    _Float16*  q16   = (_Float16*) carve((size_t)B * H * A * DH * 2);
    _Float16*  k16   = (_Float16*) carve((size_t)B * H * A * DH * 2);
    _Float16*  vT16  = (_Float16*) carve((size_t)B * H * DH * A * 2);
    _Float16*  h16   = (_Float16*) carve((size_t)M * F * 2);
    _Float16*  Wqt   = (_Float16*) carve((size_t)D * D * 2);
    _Float16*  Wkt   = (_Float16*) carve((size_t)D * D * 2);
    _Float16*  Wvt   = (_Float16*) carve((size_t)D * D * 2);
    _Float16*  W1t   = (_Float16*) carve((size_t)F * D * 2);
    _Float16*  W2t   = (_Float16*) carve((size_t)D * F * 2);
    float*     stats = (float*)    carve(16 * 4);
    float*     invsc = (float*)    carve(8 * 4);
    (void)ws_size; (void)n_in; (void)in_sizes; (void)out_size;

    // 1. weights -> f16 transposed [N,K]
    w_transpose_f16<<<(D * D + 255) / 256, 256, 0, stream>>>(Wq, Wqt, D, D);
    w_transpose_f16<<<(D * D + 255) / 256, 256, 0, stream>>>(Wk, Wkt, D, D);
    w_transpose_f16<<<(D * D + 255) / 256, 256, 0, stream>>>(Wv, Wvt, D, D);
    w_transpose_f16<<<(D * F + 255) / 256, 256, 0, stream>>>(W1, W1t, D, F);
    w_transpose_f16<<<(F * D + 255) / 256, 256, 0, stream>>>(W2, W2t, F, D);

    // 2. LN1
    ln_kernel<<<M, 256, 0, stream>>>(x, nullptr, g1, be1, xln32, xln16);

    // 3. QKV projections (WMMA GEMM, async-LDS double-buffered 128x128 tiles)
    dim3 gqkv(M / 128, D / 128);
    gemm_qkv_kernel<<<gqkv, 256, 0, stream>>>(xln16, Wqt, bq, q16, 0);
    gemm_qkv_kernel<<<gqkv, 256, 0, stream>>>(xln16, Wkt, bk, k16, 0);
    gemm_qkv_kernel<<<gqkv, 256, 0, stream>>>(xln16, Wvt, bv, vT16, 1);

    // 4. per-head score stats -> scaling
    hipMemsetAsync(stats, 0, 16 * sizeof(float), stream);
    dim3 gattn(A / 16 / 4, B * H);
    attn_stats_kernel<<<gattn, 128, 0, stream>>>(q16, k16, stats);
    attn_scale_kernel<<<1, 32, 0, stream>>>(stats, invsc);

    // 5. softmax(S/scale) @ V
    attn_softmax_pv_kernel<<<gattn, 128, 0, stream>>>(q16, k16, vT16, invsc, attn);

    // 6. residual + LN2
    ln_kernel<<<M, 256, 0, stream>>>(xln32, attn, g2, be2, xt32, xt16);

    // 7. FFN
    gemm_ffn1_kernel<<<dim3(M / 128, F / 128), 256, 0, stream>>>(xt16, W1t, bf1, h16);
    gemm_ffn2_kernel<<<dim3(M / 128, D / 128), 256, 0, stream>>>(h16, W2t, bf2, xt32, out);
}
